// ExpertChoiceMoE_29643864277425
// MI455X (gfx1250) — compile-verified, matrix operand order checked by reading, not run on hardware
//
#include <hip/hip_runtime.h>

// ---------------- problem constants ----------------
#define T_TOK 8192
#define DM    1024
#define DFF   4096
#define DFS   2048
#define NE    8
#define CAP   1228   // int(8192*1.2/8)
#define CAPR  1280   // padded capacity (multiple of 128)

typedef __bf16 bf16_t;
typedef bf16_t v16bf __attribute__((ext_vector_type(16)));
typedef bf16_t v8bf  __attribute__((ext_vector_type(8)));
typedef bf16_t v4bf  __attribute__((ext_vector_type(4)));
typedef bf16_t v2bf  __attribute__((ext_vector_type(2)));
typedef float  v8f   __attribute__((ext_vector_type(8)));
typedef float  v4f   __attribute__((ext_vector_type(4)));

// ---------------- gating: logits, top-2, softmax ----------------
__global__ __launch_bounds__(256) void gate_kernel(const float* __restrict__ x,
                                                   const float* __restrict__ gw,
                                                   int* __restrict__ idx,
                                                   float* __restrict__ probs) {
    __shared__ float sg[DM * NE];   // 32 KB
    int tid = threadIdx.x;
    for (int i = tid; i < DM * NE; i += 256) sg[i] = gw[i];
    __syncthreads();

    int t = blockIdx.x * 256 + tid;
    const float* xr = x + (size_t)t * DM;
    float acc[NE];
#pragma unroll
    for (int e = 0; e < NE; ++e) acc[e] = 0.f;
    for (int d = 0; d < DM; d += 4) {
        v4f xv = *(const v4f*)(xr + d);
#pragma unroll
        for (int c = 0; c < 4; ++c) {
#pragma unroll
            for (int e = 0; e < NE; ++e) acc[e] += xv[c] * sg[(d + c) * NE + e];
        }
    }
    // top-1 then top-2 (strict > keeps lowest index on ties, like lax.top_k)
    int i0 = 0; float m0 = acc[0];
#pragma unroll
    for (int e = 1; e < NE; ++e) if (acc[e] > m0) { m0 = acc[e]; i0 = e; }
    int i1 = -1; float m1 = -3.4e38f;
#pragma unroll
    for (int e = 0; e < NE; ++e) if (e != i0 && acc[e] > m1) { m1 = acc[e]; i1 = e; }
    float e1  = __expf(m1 - m0);
    float inv = 1.f / (1.f + e1);
    idx[2 * t]     = i0;
    idx[2 * t + 1] = i1;
    probs[2 * t]     = inv;
    probs[2 * t + 1] = e1 * inv;
}

// ---------------- routing: capacity scan + aux loss ----------------
__global__ __launch_bounds__(256) void route_kernel(const int* __restrict__ idx,
                                                    const float* __restrict__ probs,
                                                    int* __restrict__ rows,
                                                    float* __restrict__ wgt,
                                                    int* __restrict__ smap,
                                                    int* __restrict__ dropc,
                                                    float* __restrict__ aux_out) {
    __shared__ signed char sidx[2 * T_TOK];   // 16 KB
    __shared__ float loadv[NE];
    int tid = threadIdx.x;
    for (int i = tid; i < 2 * T_TOK; i += 256) {
        sidx[i] = (signed char)idx[i];
        smap[i] = -1;
    }
    for (int i = tid; i < T_TOK; i += 256) dropc[i] = 0;
    __syncthreads();

    if (tid < NE) {
        int e = tid, cnt = 0, total = 0;
        for (int t = 0; t < T_TOK; ++t) {
#pragma unroll
            for (int k = 0; k < 2; ++k) {
                if ((int)sidx[2 * t + k] == e) {
                    ++total;
                    if (cnt < CAP) {
                        rows[e * CAPR + cnt] = t;
                        wgt[e * CAPR + cnt]  = probs[2 * t + k];
                        smap[2 * t + k]      = e * CAPR + cnt;
                        ++cnt;
                    } else {
                        atomicAdd(&dropc[t], 1);   // integer: deterministic
                    }
                }
            }
        }
        for (int s = cnt; s < CAPR; ++s) { rows[e * CAPR + s] = 0; wgt[e * CAPR + s] = 0.f; }
        loadv[e] = (float)total;
    }
    __syncthreads();
    if (tid == 0) {
        float mean = 0.f;
#pragma unroll
        for (int e = 0; e < NE; ++e) mean += loadv[e];
        mean *= (1.f / NE);
        float var = 0.f;
#pragma unroll
        for (int e = 0; e < NE; ++e) { float d = loadv[e] - mean; var += d * d; }
        var *= (1.f / (NE - 1));   // ddof=1
        aux_out[0] = 0.01f * sqrtf(var) * (float)NE;
    }
}

// ---------------- WMMA fragment load from LDS ----------------
// 16-bit A layout (ISA 7.12.2): lane L (M = L&15), kb = 8*(L>>4):
//   elements 0..7  -> K = kb..kb+7        (one 16B chunk)
//   elements 8..15 -> K = 16+kb..16+kb+7  (one 16B chunk)
// B stored transposed in LDS ([n][k]) so the same pattern applies with N = L&15.
__device__ __forceinline__ v16bf frag_ld(const bf16_t* p, int kb) {
    v8bf lo = *(const v8bf*)(p + kb);
    v8bf hi = *(const v8bf*)(p + 16 + kb);
    return __builtin_shufflevector(lo, hi, 0, 1, 2, 3, 4, 5, 6, 7,
                                           8, 9, 10, 11, 12, 13, 14, 15);
}

// ---------------- tiled WMMA GEMM ----------------
// MODE 0: routed up-proj  : A = f32 x gathered via rowIdx, dual B, epi silu*mul -> bf16
// MODE 1: routed down-proj: A = bf16 h, single B, epi *rowW -> f32
// MODE 2: shared up-proj  : A = f32 x direct, dual B, epi silu*mul -> bf16
// MODE 3: shared down-proj: A = bf16 hs, single B, epi += out (f32)
#define BM 128
#define BN 128
#define BK 32
#define LDT 40   // padded LDS row stride in bf16 (80 B, multiple of 16 B)

template <int MODE>
__global__ __launch_bounds__(256) void moe_gemm(
    const float* __restrict__ Af, const bf16_t* __restrict__ Ab,
    const float* __restrict__ B1, const float* __restrict__ B2,
    const int* __restrict__ rowIdx, const float* __restrict__ rowW,
    bf16_t* __restrict__ Obf, float* __restrict__ Of,
    int M, int N, int K) {
    constexpr bool DUAL = (MODE == 0 || MODE == 2);
    constexpr bool A_BF = (MODE == 1 || MODE == 3);

    __shared__ bf16_t As[BM * LDT];
    __shared__ bf16_t Bs[BN * LDT];
    __shared__ bf16_t Cs[BN * LDT];

    const int tid  = threadIdx.x;
    const int bx   = blockIdx.x, by = blockIdx.y;
    const int lane = tid & 31, wid = tid >> 5;
    const int wm = (wid >> 2) * 64;    // wave M offset (0 / 64)
    const int wn = (wid & 3) * 32;     // wave N offset (0/32/64/96)
    const int kb = (lane >> 4) * 8;    // half-wave K base
    const int lr = lane & 15;          // row/col within subtile

    // ---- K-invariant staging indices ----
    // A (f32): 4 chunks of 4 floats; A (bf16): 2 chunks of 8 bf16
    // B: 2 chunks, each a 2(K) x 4(N) sub-block -> packed ds_store_b32 pairs
    int aRow[4], aCol[4];
    if (!A_BF) {
#pragma unroll
        for (int it = 0; it < 4; ++it) {
            int ch = tid + it * 256;
            int r = ch >> 3, c = (ch & 7) << 2;
            aCol[it] = c;
            aRow[it] = (MODE == 0) ? rowIdx[by * BM + r] : (by * BM + r);
        }
    } else {
#pragma unroll
        for (int it = 0; it < 2; ++it) {
            int ch = tid + it * 256;
            aRow[it] = by * BM + (ch >> 2);
            aCol[it] = (ch & 3) << 3;
        }
    }
    int bK2[2], bN4[2];
#pragma unroll
    for (int it = 0; it < 2; ++it) {
        int ch = tid + it * 256;
        bK2[it] = (ch >> 5) << 1;          // even K within tile
        bN4[it] = (ch & 31) << 2;          // N within tile
    }

    // ---- register staging buffers (double-buffer against WMMA) ----
    v4f  aR[4];
    v8bf aRb[2];
    v4f  b1R[2][2], b2R[2][2];

    auto fetch = [&](int k0) {
        if (!A_BF) {
#pragma unroll
            for (int it = 0; it < 4; ++it)
                aR[it] = *(const v4f*)(Af + (size_t)aRow[it] * K + k0 + aCol[it]);
        } else {
#pragma unroll
            for (int it = 0; it < 2; ++it)
                aRb[it] = *(const v8bf*)(Ab + (size_t)aRow[it] * K + k0 + aCol[it]);
        }
#pragma unroll
        for (int it = 0; it < 2; ++it) {
            size_t go = (size_t)(k0 + bK2[it]) * N + bx * BN + bN4[it];
            b1R[it][0] = *(const v4f*)(B1 + go);
            b1R[it][1] = *(const v4f*)(B1 + go + N);
            if (DUAL) {
                b2R[it][0] = *(const v4f*)(B2 + go);
                b2R[it][1] = *(const v4f*)(B2 + go + N);
            }
            if (it == 0 && k0 + BK < K)
                __builtin_prefetch(B1 + go + (size_t)BK * N, 0, 0);
        }
    };
    auto lds_store = [&]() {
        if (!A_BF) {
#pragma unroll
            for (int it = 0; it < 4; ++it) {
                int ch = tid + it * 256;
                int r = ch >> 3;
                v4bf t = { (bf16_t)aR[it][0], (bf16_t)aR[it][1],
                           (bf16_t)aR[it][2], (bf16_t)aR[it][3] };
                *(v4bf*)&As[r * LDT + aCol[it]] = t;       // ds_store_b64
            }
        } else {
#pragma unroll
            for (int it = 0; it < 2; ++it) {
                int ch = tid + it * 256;
                int r = ch >> 2;
                *(v8bf*)&As[r * LDT + aCol[it]] = aRb[it]; // ds_store_b128
            }
        }
#pragma unroll
        for (int it = 0; it < 2; ++it) {
#pragma unroll
            for (int i = 0; i < 4; ++i) {
                v2bf p = { (bf16_t)b1R[it][0][i], (bf16_t)b1R[it][1][i] };
                *(v2bf*)&Bs[(bN4[it] + i) * LDT + bK2[it]] = p;   // ds_store_b32
                if (DUAL) {
                    v2bf q = { (bf16_t)b2R[it][0][i], (bf16_t)b2R[it][1][i] };
                    *(v2bf*)&Cs[(bN4[it] + i) * LDT + bK2[it]] = q;
                }
            }
        }
    };

    v8f acc1[4][2], acc2[4][2];
    const v8f vz = {0.f, 0.f, 0.f, 0.f, 0.f, 0.f, 0.f, 0.f};
#pragma unroll
    for (int i = 0; i < 4; ++i)
#pragma unroll
        for (int j = 0; j < 2; ++j) { acc1[i][j] = vz; acc2[i][j] = vz; }

    fetch(0);
    for (int k0 = 0; k0 < K; k0 += BK) {
        lds_store();
        __syncthreads();
        if (k0 + BK < K) fetch(k0 + BK);   // overlap next-tile loads with WMMA

        v16bf af[4], bfr[2], cfr[2];
#pragma unroll
        for (int i = 0; i < 4; ++i) af[i] = frag_ld(&As[(wm + i * 16 + lr) * LDT], kb);
#pragma unroll
        for (int j = 0; j < 2; ++j) {
            bfr[j] = frag_ld(&Bs[(wn + j * 16 + lr) * LDT], kb);
            if (DUAL) cfr[j] = frag_ld(&Cs[(wn + j * 16 + lr) * LDT], kb);
        }
#pragma unroll
        for (int i = 0; i < 4; ++i) {
#pragma unroll
            for (int j = 0; j < 2; ++j) {
                acc1[i][j] = __builtin_amdgcn_wmma_f32_16x16x32_bf16(
                    false, af[i], false, bfr[j], (short)0, acc1[i][j], false, false);
                if (DUAL)
                    acc2[i][j] = __builtin_amdgcn_wmma_f32_16x16x32_bf16(
                        false, af[i], false, cfr[j], (short)0, acc2[i][j], false, false);
            }
        }
        __syncthreads();
    }

    // ---- epilogue ----
    // C/D layout: vgpr r -> M = r + 8*(lane>>4); N = lane&15
    const int hi = lane >> 4;
#pragma unroll
    for (int i = 0; i < 4; ++i) {
#pragma unroll
        for (int j = 0; j < 2; ++j) {
#pragma unroll
            for (int r = 0; r < 8; ++r) {
                int gm = by * BM + wm + i * 16 + hi * 8 + r;
                int gn = bx * BN + wn + j * 16 + lr;
                if (MODE == 0 || MODE == 2) {
                    float g = acc1[i][j][r];
                    float s = g / (1.f + __expf(-g));       // silu
                    Obf[(size_t)gm * N + gn] = (bf16_t)(s * acc2[i][j][r]);
                } else if (MODE == 1) {
                    Of[(size_t)gm * N + gn] = acc1[i][j][r] * rowW[gm];
                } else {
                    Of[(size_t)gm * N + gn] += acc1[i][j][r];
                }
            }
        }
    }
}

// ---------------- combine: drop residual + routed scatter ----------------
__global__ __launch_bounds__(256) void combine_kernel(const float* __restrict__ x,
                                                      const int* __restrict__ dropc,
                                                      const int* __restrict__ smap,
                                                      const float* __restrict__ eo,
                                                      float* __restrict__ out) {
    size_t i = (size_t)blockIdx.x * 256 + threadIdx.x;
    int t = (int)(i >> 10);         // / DM
    int d = (int)(i & (DM - 1));
    float acc = (float)dropc[t] * x[i];
    int s0 = smap[2 * t], s1 = smap[2 * t + 1];
    if (s0 >= 0) acc += eo[(size_t)s0 * DM + d];
    if (s1 >= 0) acc += eo[(size_t)s1 * DM + d];
    out[i] = acc;
}

// ---------------- launcher ----------------
extern "C" void kernel_launch(void* const* d_in, const int* in_sizes, int n_in,
                              void* d_out, int out_size, void* d_ws, size_t ws_size,
                              hipStream_t stream) {
    const float* x   = (const float*)d_in[0];
    const float* gw  = (const float*)d_in[1];
    const float* w1  = (const float*)d_in[2];
    const float* w2  = (const float*)d_in[3];
    const float* w3  = (const float*)d_in[4];
    const float* sw1 = (const float*)d_in[5];
    const float* sw2 = (const float*)d_in[6];
    const float* sw3 = (const float*)d_in[7];
    float* out = (float*)d_out;

    char* ws = (char*)d_ws;
    int*    idx   = (int*)   (ws + 0);          //  64 KB
    float*  probs = (float*) (ws + 65536);      //  64 KB
    int*    rows  = (int*)   (ws + 131072);     //  40 KB
    float*  wgt   = (float*) (ws + 172032);     //  40 KB
    int*    smap  = (int*)   (ws + 212992);     //  64 KB
    int*    dropc = (int*)   (ws + 278528);     //  32 KB
    bf16_t* h     = (bf16_t*)(ws + 311296);     //  10.0 MB (per-expert, reused)
    float*  eo    = (float*) (ws + 10797056);   //  40.0 MB (all experts)
    bf16_t* hs    = (bf16_t*)(ws + 52740096);   //  32.0 MB (per shared-expert, reused)
    // total ws use: 86,294,528 bytes

    gate_kernel<<<T_TOK / 256, 256, 0, stream>>>(x, gw, idx, probs);
    route_kernel<<<1, 256, 0, stream>>>(idx, probs, rows, wgt, smap, dropc,
                                        out + (size_t)T_TOK * DM);

    for (int e = 0; e < NE; ++e) {
        moe_gemm<0><<<dim3(DFF / BN, CAPR / BM), 256, 0, stream>>>(
            x, nullptr,
            w1 + (size_t)e * DM * DFF, w2 + (size_t)e * DM * DFF,
            rows + e * CAPR, nullptr,
            h, nullptr, CAPR, DFF, DM);
        moe_gemm<1><<<dim3(DM / BN, CAPR / BM), 256, 0, stream>>>(
            nullptr, h,
            w3 + (size_t)e * DFF * DM, nullptr,
            nullptr, wgt + e * CAPR,
            nullptr, eo + (size_t)e * CAPR * DM, CAPR, DM, DFF);
    }

    combine_kernel<<<(T_TOK * DM) / 256, 256, 0, stream>>>(x, dropc, smap, eo, out);

    for (int se = 0; se < 2; ++se) {
        moe_gemm<2><<<dim3(DFS / BN, T_TOK / BM), 256, 0, stream>>>(
            x, nullptr,
            sw1 + (size_t)se * DM * DFS, sw2 + (size_t)se * DM * DFS,
            nullptr, nullptr,
            hs, nullptr, T_TOK, DFS, DM);
        moe_gemm<3><<<dim3(DM / BN, T_TOK / BM), 256, 0, stream>>>(
            nullptr, hs,
            sw3 + (size_t)se * DFS * DM, nullptr,
            nullptr, nullptr,
            nullptr, out, T_TOK, DM, DFS);
    }
}
